// GraphTransformer_73443940762152
// MI455X (gfx1250) — compile-verified, matrix operand order checked by reading, not run on hardware
//
#include <hip/hip_runtime.h>

typedef __attribute__((ext_vector_type(16))) __bf16 v16bf;
typedef __attribute__((ext_vector_type(8)))  float  v8f;

#define BB 2
#define NN 2000
#define KKN 30
#define CC 128
#define HH 4
#define DD 32
#define NEG_INF_F (-3.4028234663852886e+38f)

// ---------------------------------------------------------------- helpers
__device__ __forceinline__ v8f wmma_bf16(v16bf a, v16bf b, v8f c) {
  // v_wmma_f32_16x16x32_bf16: D = A(16x32) * B(32x16) + C
  return __builtin_amdgcn_wmma_f32_16x16x32_bf16(false, a, false, b, (short)0, c,
                                                 false, false);
}

__device__ __forceinline__ float silu_f(float x) { return x / (1.f + __expf(-x)); }

// pack 8 fp32 @p0 and 8 fp32 @p1 into one bf16 A/B fragment register set
__device__ __forceinline__ v16bf pack16_f32(const float* p0, const float* p1) {
  float4 a0 = *(const float4*)p0;
  float4 a1 = *(const float4*)(p0 + 4);
  float4 b0 = *(const float4*)p1;
  float4 b1 = *(const float4*)(p1 + 4);
  union { v16bf v; __bf16 e[16]; } u;
  u.e[0]=(__bf16)a0.x; u.e[1]=(__bf16)a0.y; u.e[2]=(__bf16)a0.z; u.e[3]=(__bf16)a0.w;
  u.e[4]=(__bf16)a1.x; u.e[5]=(__bf16)a1.y; u.e[6]=(__bf16)a1.z; u.e[7]=(__bf16)a1.w;
  u.e[8]=(__bf16)b0.x; u.e[9]=(__bf16)b0.y; u.e[10]=(__bf16)b0.z; u.e[11]=(__bf16)b0.w;
  u.e[12]=(__bf16)b1.x; u.e[13]=(__bf16)b1.y; u.e[14]=(__bf16)b1.z; u.e[15]=(__bf16)b1.w;
  return u.v;
}

__device__ __forceinline__ v16bf pack16_lds(const __bf16* p0, const __bf16* p1) {
  union { v16bf v; __bf16 e[16]; } u;
#pragma unroll
  for (int i = 0; i < 8; ++i) u.e[i] = p0[i];
#pragma unroll
  for (int i = 0; i < 8; ++i) u.e[8 + i] = p1[i];
  return u.v;
}

__device__ __forceinline__ float block_reduce_sum256(float v, float* red) {
  int t = threadIdx.x;
  red[t] = v;
  __syncthreads();
#pragma unroll
  for (int s = 128; s > 0; s >>= 1) {
    if (t < s) red[t] += red[t + s];
    __syncthreads();
  }
  float r = red[0];
  __syncthreads();
  return r;
}

__device__ __forceinline__ float block_reduce_sum128(float v, float* red) {
  int t = threadIdx.x;
  red[t] = v;
  __syncthreads();
#pragma unroll
  for (int s = 64; s > 0; s >>= 1) {
    if (t < s) red[t] += red[t + s];
    __syncthreads();
  }
  float r = red[0];
  __syncthreads();
  return r;
}

// ---------------------------------------------------------------- K0: weight repack
// Converts fp32 row-major (Kd x Nd) weights into bf16 WMMA B-fragment layout:
// frag ft = kt*NT + nt ; element (lane, j):
//   n = nt*16 + (lane&15)
//   k = kt*32 + (lane>=16 ? 8 : 0) + (j>=8 ? 16 : 0) + (j&7)
// Matches the ISA 16-bit operand VGPR layout (CDNA5 ISA 7.12.2).
__global__ __launch_bounds__(256) void prep_kernel(
    const float* __restrict__ nk, const float* __restrict__ nv,
    const float* __restrict__ em1, const float* __restrict__ em2,
    const float* __restrict__ f1, const float* __restrict__ f2,
    __bf16* __restrict__ frags) {
  int gid = blockIdx.x * 256 + threadIdx.x;  // < 512*512
  int fr = gid >> 9;
  int within = gid & 511;
  int lane = within >> 4;
  int j = within & 15;
  const float* W;
  int Nd, lf;
  if (fr < 64)       { W = nk;  Nd = 128; lf = fr; }
  else if (fr < 128) { W = nv;  Nd = 128; lf = fr - 64; }
  else if (fr < 224) { W = em1; Nd = 128; lf = fr - 128; }
  else if (fr < 256) { W = em2; Nd = 128; lf = fr - 224; }
  else if (fr < 384) { W = f1;  Nd = 512; lf = fr - 256; }
  else               { W = f2;  Nd = 128; lf = fr - 384; }
  int NT = Nd >> 4;
  int kt = lf / NT, nt = lf % NT;
  int n = nt * 16 + (lane & 15);
  int k = kt * 32 + ((lane >> 4) << 3) + ((j >> 3) << 4) + (j & 7);
  frags[gid] = (__bf16)W[k * Nd + n];
}

// ---------------------------------------------------------------- K1: cross-attn constant
// L=1 => softmax over length-1 axis == 1 => x = v2 broadcast. xo = v2 @ co_w + co_b.
__global__ __launch_bounds__(128) void xo_kernel(
    const float* __restrict__ ligand, const float* __restrict__ cv_w,
    const float* __restrict__ cv_b, const float* __restrict__ co_w,
    const float* __restrict__ co_b, float* __restrict__ xo) {
  __shared__ float lrow[128];
  __shared__ float v2[512];
  int b = blockIdx.x, t = threadIdx.x;
  lrow[t] = ligand[b * CC + t];
  __syncthreads();
  for (int jj = t; jj < 512; jj += 128) {
    float s = cv_b[jj];
    for (int d = 0; d < 128; ++d) s += lrow[d] * cv_w[d * 512 + jj];
    v2[jj] = s;
  }
  __syncthreads();
  float s = co_b[t];
  for (int jj = 0; jj < 512; ++jj) s += v2[jj] * co_w[jj * CC + t];
  xo[b * CC + t] = s;
}

// ---------------------------------------------------------------- K2a: neighbor attention (WMMA)
__global__ __launch_bounds__(256) void attn_kernel(
    const float* __restrict__ node, const float* __restrict__ edge,
    const int* __restrict__ edge_index, const float* __restrict__ mask_attend,
    const __bf16* __restrict__ frag_nk, const __bf16* __restrict__ frag_nv,
    const float* __restrict__ nq_w, const float* __restrict__ no_w,
    const float* __restrict__ ln0_g, const float* __restrict__ ln0_b,
    const float* __restrict__ ln1_g, const float* __restrict__ ln1_b,
    const float* __restrict__ xo, float* __restrict__ node_mid) {
  __shared__ __bf16 aev[32 * 264];   // h_EV tile, bf16, padded stride
  __shared__ float kv[32 * 260];     // k (cols 0..127) and v (cols 128..255)
  __shared__ float nrow[128], qv[128], dh[128], att[128];
  __shared__ float red[256];

  const int t = threadIdx.x;
  const int bn = blockIdx.x;
  const int b = bn / NN;
  const int wave = t >> 5, lane = t & 31;
  const int hi = (lane >> 4) & 1, lm = lane & 15;

  if (t < 128) nrow[t] = node[(size_t)bn * CC + t];
  const int* eidx = edge_index + (size_t)bn * KKN;
  for (int e = t; e < 32 * 256; e += 256) {
    int row = e >> 8, c = e & 255;
    float v = 0.f;
    if (row < KKN) {
      v = (c < CC) ? edge[((size_t)bn * KKN + row) * CC + c]
                   : node[((size_t)b * NN + eidx[row]) * CC + (c - CC)];
    }
    aev[row * 264 + c] = (__bf16)v;
  }
  __syncthreads();

  if (t < 128) {  // q = node_row @ nq (1x128 dot products, scalar path)
    float s = 0.f;
    for (int d = 0; d < CC; ++d) s += nrow[d] * nq_w[d * CC + t];
    qv[t] = s;
  }

  // 32 output tiles (k:16, v:16); 4 per wave; 8 k-steps each => 256 WMMA/node
#pragma unroll
  for (int q4 = 0; q4 < 4; ++q4) {
    int tt = wave * 4 + q4;
    int g = tt >> 4;
    int rt = (tt >> 3) & 1;
    int ct = tt & 7;
    const __bf16* fb = g ? frag_nv : frag_nk;
    const int arow = rt * 16 + lm;
    v8f acc = {0.f, 0.f, 0.f, 0.f, 0.f, 0.f, 0.f, 0.f};
#pragma unroll
    for (int kt = 0; kt < 8; ++kt) {
      int c0 = kt * 32 + hi * 8;
      v16bf a = pack16_lds(&aev[arow * 264 + c0], &aev[arow * 264 + c0 + 16]);
      v16bf bf = ((const v16bf*)fb)[(kt * 8 + ct) * 32 + lane];
      acc = wmma_bf16(a, bf, acc);
    }
#pragma unroll
    for (int v = 0; v < 8; ++v)
      kv[(rt * 16 + hi * 8 + v) * 260 + g * 128 + ct * 16 + lm] = acc[v];
  }
  __syncthreads();

  // logits + masked softmax: wave h handles head h via lane shuffles
  if (t < 128) {
    int h = t >> 5, j = t & 31;
    float mval = 0.f, l = NEG_INF_F;
    if (j < KKN) {
      mval = mask_attend[(size_t)bn * KKN + j];
      float s = 0.f;
#pragma unroll
      for (int d = 0; d < DD; ++d) s += qv[h * DD + d] * kv[j * 260 + h * DD + d];
      l = (mval > 0.f) ? s * 0.17677669529663687f : NEG_INF_F;
    }
    float mx = l;
#pragma unroll
    for (int off = 16; off > 0; off >>= 1) mx = fmaxf(mx, __shfl_xor(mx, off, 32));
    float ex = (j < KKN && mval > 0.f) ? __expf(l - mx) : 0.f;
    float sm = ex;
#pragma unroll
    for (int off = 16; off > 0; off >>= 1) sm += __shfl_xor(sm, off, 32);
    att[t] = (sm > 0.f) ? (ex / sm) * mval : 0.f;
  }
  __syncthreads();

  if (t < 128) {  // dh = att @ v
    int h = t >> 5, d = t & 31;
    float s = 0.f;
    for (int j = 0; j < KKN; ++j) s += att[h * 32 + j] * kv[j * 260 + 128 + h * DD + d];
    dh[t] = s;
  }
  __syncthreads();

  float x1 = 0.f;
  if (t < 128) {  // o = dh @ no, residual
    float s = 0.f;
    for (int d = 0; d < CC; ++d) s += dh[d] * no_w[d * CC + t];
    x1 = nrow[t] + s;
  }
  float s1 = block_reduce_sum256((t < 128) ? x1 : 0.f, red);
  float s2 = block_reduce_sum256((t < 128) ? x1 * x1 : 0.f, red);
  float mean = s1 / 128.f;
  float var = s2 / 128.f - mean * mean;
  float x2 = 0.f;
  if (t < 128) {
    float y1 = (x1 - mean) * rsqrtf(var + 1e-5f) * ln0_g[t] + ln0_b[t];
    x2 = y1 + xo[b * CC + t];
  }
  float s3 = block_reduce_sum256((t < 128) ? x2 : 0.f, red);
  float s4 = block_reduce_sum256((t < 128) ? x2 * x2 : 0.f, red);
  float mean2 = s3 / 128.f;
  float var2 = s4 / 128.f - mean2 * mean2;
  if (t < 128)
    node_mid[(size_t)bn * CC + t] =
        (x2 - mean2) * rsqrtf(var2 + 1e-5f) * ln1_g[t] + ln1_b[t];
}

// ---------------------------------------------------------------- K2b: FFN (WMMA) + LN2 + mask
__global__ __launch_bounds__(64) void ffn_kernel(
    const float* __restrict__ node_mid, const float* __restrict__ mask,
    const __bf16* __restrict__ frag_f1, const __bf16* __restrict__ frag_f2,
    const float* __restrict__ f1_b, const float* __restrict__ f2_b,
    const float* __restrict__ ln2_g, const float* __restrict__ ln2_b,
    float* __restrict__ out_node) {
  __shared__ __bf16 h1[2][16 * 520];
  const int wave = threadIdx.x >> 5, lane = threadIdx.x & 31;
  const int tile = blockIdx.x * 2 + wave;  // grid 125 -> tile < 250 (exact)
  const int lm = lane & 15, hi = (lane >> 4) & 1;
  const float* arow = node_mid + ((size_t)tile * 16 + lm) * CC;

  v16bf afr[4];
#pragma unroll
  for (int kt = 0; kt < 4; ++kt) {
    int c0 = kt * 32 + hi * 8;
    afr[kt] = pack16_f32(arow + c0, arow + c0 + 16);
  }
  __bf16* my = h1[wave];
  for (int ct = 0; ct < 32; ++ct) {
    v8f acc = {0.f, 0.f, 0.f, 0.f, 0.f, 0.f, 0.f, 0.f};
#pragma unroll
    for (int kt = 0; kt < 4; ++kt) {
      v16bf bf = ((const v16bf*)frag_f1)[(kt * 32 + ct) * 32 + lane];
      acc = wmma_bf16(afr[kt], bf, acc);
    }
#pragma unroll
    for (int v = 0; v < 8; ++v) {
      int col = ct * 16 + lm;
      float x = acc[v] + f1_b[col];
      my[(v + hi * 8) * 520 + col] = (__bf16)fmaxf(x, 0.f);
    }
  }
  __syncthreads();

  v8f acc2[8];
#pragma unroll
  for (int ct = 0; ct < 8; ++ct) acc2[ct] = (v8f){0.f,0.f,0.f,0.f,0.f,0.f,0.f,0.f};
  for (int kt = 0; kt < 16; ++kt) {
    int c0 = kt * 32 + hi * 8;
    v16bf a = pack16_lds(&my[lm * 520 + c0], &my[lm * 520 + c0 + 16]);
#pragma unroll
    for (int ct = 0; ct < 8; ++ct) {
      v16bf bf = ((const v16bf*)frag_f2)[(kt * 8 + ct) * 32 + lane];
      acc2[ct] = wmma_bf16(a, bf, acc2[ct]);
    }
  }

  // epilogue: residual + LN2 + mask, fully in registers (shuffle reduction)
  float val[8][8], rs[8], rq[8];
#pragma unroll
  for (int v = 0; v < 8; ++v) { rs[v] = 0.f; rq[v] = 0.f; }
#pragma unroll
  for (int ct = 0; ct < 8; ++ct) {
#pragma unroll
    for (int v = 0; v < 8; ++v) {
      int col = ct * 16 + lm;
      int grow = tile * 16 + v + hi * 8;
      float x = acc2[ct][v] + f2_b[col] + node_mid[(size_t)grow * CC + col];
      val[ct][v] = x;
      rs[v] += x; rq[v] += x * x;
    }
  }
#pragma unroll
  for (int v = 0; v < 8; ++v) {
#pragma unroll
    for (int off = 8; off > 0; off >>= 1) {
      rs[v] += __shfl_xor(rs[v], off, 32);
      rq[v] += __shfl_xor(rq[v], off, 32);
    }
  }
#pragma unroll
  for (int ct = 0; ct < 8; ++ct) {
#pragma unroll
    for (int v = 0; v < 8; ++v) {
      int col = ct * 16 + lm;
      int grow = tile * 16 + v + hi * 8;
      float mean = rs[v] * (1.f / 128.f);
      float var = rq[v] * (1.f / 128.f) - mean * mean;
      float y = (val[ct][v] - mean) * rsqrtf(var + 1e-5f) * ln2_g[col] + ln2_b[col];
      out_node[(size_t)grow * CC + col] = y * mask[grow];
    }
  }
}

// ---------------------------------------------------------------- K3: masked mean / max pooling
__global__ __launch_bounds__(256) void pool_kernel(
    const float* __restrict__ out_node, const float* __restrict__ mask,
    float* __restrict__ sums, float* __restrict__ maxs, float* __restrict__ msum) {
  int t = threadIdx.x;
  int b = t >> 7, c = t & 127;
  float s = 0.f, mx = NEG_INF_F;
  for (int n = 0; n < NN; ++n) {
    float v = out_node[((size_t)b * NN + n) * CC + c];
    s += v;
    mx = fmaxf(mx, v);
  }
  sums[t] = s;
  maxs[t] = mx;
  if (c == 0) {
    float ms = 0.f;
    for (int n = 0; n < NN; ++n) ms += mask[b * NN + n];
    msum[b] = ms;
  }
}

// ---------------------------------------------------------------- K4: ligand update (tiny)
__global__ __launch_bounds__(128) void ligand_kernel(
    const float* __restrict__ ligand, const float* __restrict__ sums,
    const float* __restrict__ maxs, const float* __restrict__ msum,
    const float* __restrict__ cm1_w, const float* __restrict__ cm1_b,
    const float* __restrict__ cm2_w, const float* __restrict__ cm2_b,
    const float* __restrict__ lnc_g, const float* __restrict__ lnc_b,
    float* __restrict__ out_lig) {
  __shared__ float hL[384], t1[128], red[128];
  int b = blockIdx.x, t = threadIdx.x;
  float lg = ligand[b * CC + t];
  hL[t] = sums[b * CC + t] / msum[b];
  hL[128 + t] = maxs[b * CC + t];
  hL[256 + t] = lg;
  __syncthreads();
  float s = cm1_b[t];
  for (int d = 0; d < 384; ++d) s += hL[d] * cm1_w[d * CC + t];
  t1[t] = silu_f(s);
  __syncthreads();
  float s2 = cm2_b[t];
  for (int d = 0; d < 128; ++d) s2 += t1[d] * cm2_w[d * CC + t];
  float x = lg + silu_f(s2);
  float s1r = block_reduce_sum128(x, red);
  float s2r = block_reduce_sum128(x * x, red);
  float mean = s1r / 128.f, var = s2r / 128.f - mean * mean;
  out_lig[b * CC + t] = (x - mean) * rsqrtf(var + 1e-5f) * lnc_g[t] + lnc_b[t];
}

// ---------------------------------------------------------------- K5: edge update (WMMA, big)
__global__ __launch_bounds__(128) void edge_kernel(
    const float* __restrict__ node_fin, const float* __restrict__ edge_in,
    const int* __restrict__ edge_index,
    const __bf16* __restrict__ frag_em1, const __bf16* __restrict__ frag_em2,
    const float* __restrict__ em1_b, const float* __restrict__ em2_b,
    const float* __restrict__ lne_g, const float* __restrict__ lne_b,
    float* __restrict__ out_edge) {
  __shared__ __bf16 h1[4][16 * 136];
  const int wave = threadIdx.x >> 5, lane = threadIdx.x & 31;
  const int tile = blockIdx.x * 4 + wave;  // grid 1875 -> tile < 7500 (exact)
  const int lm = lane & 15, hi = (lane >> 4) & 1;
  const int r = tile * 16 + lm;
  const int b = r / (NN * KKN);
  const int rem = r % (NN * KKN);
  const int n = rem / KKN;
  const int jn = edge_index[r];
  const float* s0 = node_fin + ((size_t)b * NN + n) * CC;   // node_i
  const float* s1 = edge_in + (size_t)r * CC;               // edge
  const float* s2 = node_fin + ((size_t)b * NN + jn) * CC;  // node_j

  // GEMM1: h_VE(16x384) @ em1(384x128); A frags assembled straight from the
  // 3-segment concat per lane (chunks are 8-aligned, never straddle segments)
  v8f acc[8];
#pragma unroll
  for (int ct = 0; ct < 8; ++ct) acc[ct] = (v8f){0.f,0.f,0.f,0.f,0.f,0.f,0.f,0.f};
  for (int kt = 0; kt < 12; ++kt) {
    int c0 = kt * 32 + hi * 8;
    int c1 = c0 + 16;
    const float* p0 = (c0 < 128 ? s0 : (c0 < 256 ? s1 : s2)) + (c0 & 127);
    const float* p1 = (c1 < 128 ? s0 : (c1 < 256 ? s1 : s2)) + (c1 & 127);
    v16bf a = pack16_f32(p0, p1);
#pragma unroll
    for (int ct = 0; ct < 8; ++ct) {
      v16bf bf = ((const v16bf*)frag_em1)[(kt * 8 + ct) * 32 + lane];
      acc[ct] = wmma_bf16(a, bf, acc[ct]);
    }
  }
  __bf16* my = h1[wave];
#pragma unroll
  for (int ct = 0; ct < 8; ++ct) {
#pragma unroll
    for (int v = 0; v < 8; ++v) {
      int col = ct * 16 + lm;
      float x = acc[ct][v] + em1_b[col];
      my[(v + hi * 8) * 136 + col] = (__bf16)silu_f(x);
    }
  }
  __syncthreads();

  // GEMM2: h1(16x128) @ em2(128x128)
  v8f acc2[8];
#pragma unroll
  for (int ct = 0; ct < 8; ++ct) acc2[ct] = (v8f){0.f,0.f,0.f,0.f,0.f,0.f,0.f,0.f};
  for (int kt = 0; kt < 4; ++kt) {
    int c0 = kt * 32 + hi * 8;
    v16bf a = pack16_lds(&my[lm * 136 + c0], &my[lm * 136 + c0 + 16]);
#pragma unroll
    for (int ct = 0; ct < 8; ++ct) {
      v16bf bf = ((const v16bf*)frag_em2)[(kt * 8 + ct) * 32 + lane];
      acc2[ct] = wmma_bf16(a, bf, acc2[ct]);
    }
  }

  // epilogue: silu + residual + LN(lne), via shuffle reductions
  float val[8][8], rs[8], rq[8];
#pragma unroll
  for (int v = 0; v < 8; ++v) { rs[v] = 0.f; rq[v] = 0.f; }
#pragma unroll
  for (int ct = 0; ct < 8; ++ct) {
#pragma unroll
    for (int v = 0; v < 8; ++v) {
      int col = ct * 16 + lm;
      int er = tile * 16 + v + hi * 8;
      float x = silu_f(acc2[ct][v] + em2_b[col]) + edge_in[(size_t)er * CC + col];
      val[ct][v] = x;
      rs[v] += x; rq[v] += x * x;
    }
  }
#pragma unroll
  for (int v = 0; v < 8; ++v) {
#pragma unroll
    for (int off = 8; off > 0; off >>= 1) {
      rs[v] += __shfl_xor(rs[v], off, 32);
      rq[v] += __shfl_xor(rq[v], off, 32);
    }
  }
#pragma unroll
  for (int ct = 0; ct < 8; ++ct) {
#pragma unroll
    for (int v = 0; v < 8; ++v) {
      int col = ct * 16 + lm;
      int er = tile * 16 + v + hi * 8;
      float mean = rs[v] * (1.f / 128.f);
      float var = rq[v] * (1.f / 128.f) - mean * mean;
      out_edge[(size_t)er * CC + col] =
          (val[ct][v] - mean) * rsqrtf(var + 1e-5f) * lne_g[col] + lne_b[col];
    }
  }
}

// ---------------------------------------------------------------- launcher
extern "C" void kernel_launch(void* const* d_in, const int* in_sizes, int n_in,
                              void* d_out, int out_size, void* d_ws, size_t ws_size,
                              hipStream_t stream) {
  (void)in_sizes; (void)n_in; (void)out_size; (void)ws_size;
  const float* node        = (const float*)d_in[0];
  const float* edge        = (const float*)d_in[1];
  const int*   edge_index  = (const int*)d_in[2];
  const float* ligand      = (const float*)d_in[3];
  const float* mask        = (const float*)d_in[4];
  const float* mask_attend = (const float*)d_in[5];
  // params in setup_inputs() insertion order
  const float* nq    = (const float*)d_in[6];
  const float* nk    = (const float*)d_in[7];
  const float* nv    = (const float*)d_in[8];
  const float* no_w  = (const float*)d_in[9];
  const float* cv_w  = (const float*)d_in[14];
  const float* cv_b  = (const float*)d_in[15];
  const float* co_w  = (const float*)d_in[16];
  const float* co_b  = (const float*)d_in[17];
  const float* f1_w  = (const float*)d_in[18];
  const float* f1_b  = (const float*)d_in[19];
  const float* f2_w  = (const float*)d_in[20];
  const float* f2_b  = (const float*)d_in[21];
  const float* ln0_g = (const float*)d_in[22];
  const float* ln0_b = (const float*)d_in[23];
  const float* ln1_g = (const float*)d_in[24];
  const float* ln1_b = (const float*)d_in[25];
  const float* ln2_g = (const float*)d_in[26];
  const float* ln2_b = (const float*)d_in[27];
  const float* em1_w = (const float*)d_in[28];
  const float* em1_b = (const float*)d_in[29];
  const float* em2_w = (const float*)d_in[30];
  const float* em2_b = (const float*)d_in[31];
  const float* lne_g = (const float*)d_in[32];
  const float* lne_b = (const float*)d_in[33];
  const float* cm1_w = (const float*)d_in[34];
  const float* cm1_b = (const float*)d_in[35];
  const float* cm2_w = (const float*)d_in[36];
  const float* cm2_b = (const float*)d_in[37];
  const float* lnc_g = (const float*)d_in[38];
  const float* lnc_b = (const float*)d_in[39];

  // workspace layout
  __bf16* frags    = (__bf16*)d_ws;            // 512 frags * 512 elems (bf16)
  __bf16* frag_nk  = frags;
  __bf16* frag_nv  = frags + 64 * 512;
  __bf16* frag_em1 = frags + 128 * 512;
  __bf16* frag_em2 = frags + 224 * 512;
  __bf16* frag_f1  = frags + 256 * 512;
  __bf16* frag_f2  = frags + 384 * 512;
  float* node_mid = (float*)((char*)d_ws + 512 * 512 * 2);
  float* xo   = node_mid + (size_t)BB * NN * CC;
  float* sums = xo + BB * CC;
  float* maxs = sums + BB * CC;
  float* msum = maxs + BB * CC;

  float* out_node = (float*)d_out;
  float* out_edge = out_node + (size_t)BB * NN * CC;
  float* out_lig  = out_edge + (size_t)BB * NN * KKN * CC;

  prep_kernel<<<1024, 256, 0, stream>>>(nk, nv, em1_w, em2_w, f1_w, f2_w, frags);
  xo_kernel<<<BB, 128, 0, stream>>>(ligand, cv_w, cv_b, co_w, co_b, xo);
  attn_kernel<<<BB * NN, 256, 0, stream>>>(node, edge, edge_index, mask_attend,
                                           frag_nk, frag_nv, nq, no_w, ln0_g, ln0_b,
                                           ln1_g, ln1_b, xo, node_mid);
  ffn_kernel<<<(BB * NN / 16) / 2, 64, 0, stream>>>(node_mid, mask, frag_f1, frag_f2,
                                                    f1_b, f2_b, ln2_g, ln2_b, out_node);
  pool_kernel<<<1, 256, 0, stream>>>(out_node, mask, sums, maxs, msum);
  ligand_kernel<<<BB, 128, 0, stream>>>(ligand, sums, maxs, msum, cm1_w, cm1_b,
                                        cm2_w, cm2_b, lnc_g, lnc_b, out_lig);
  edge_kernel<<<(BB * NN * KKN / 16) / 4, 128, 0, stream>>>(
      out_node, edge, edge_index, frag_em1, frag_em2, em1_b, em2_b, lne_g, lne_b,
      out_edge);
}
